// Net_17738214932546
// MI455X (gfx1250) — compile-verified, matrix operand order checked by reading, not run on hardware
//
#include <hip/hip_runtime.h>

// ---------------------------------------------------------------------------
// Problem constants
// ---------------------------------------------------------------------------
#define NTOT  16384   // USERNUM + ITEMNUM
#define HALF  8192
#define DIMF  256

// GEMM tiling: block 256 thr = 8 waves (4x2), wave owns 32x32 = 2x2 WMMA tiles
#define TBM 128
#define TBN 64
#define TBK 32
#define LDSA 40       // padded LDS row stride (bf16 elems) -> 80B rows
#define LDSB 40

typedef __attribute__((ext_vector_type(16))) __bf16 v16bf;
typedef __attribute__((ext_vector_type(8)))  __bf16 v8bf;
typedef __attribute__((ext_vector_type(4)))  __bf16 v4bf;
typedef __attribute__((ext_vector_type(8)))  float  v8f;

__device__ __forceinline__ __bf16 f2bf(float f) {
    union { float f; unsigned u; } v; v.f = f;
    unsigned r = v.u + 0x7FFFu + ((v.u >> 16) & 1u);   // round-to-nearest-even
    union { unsigned short s; __bf16 b; } o; o.s = (unsigned short)(r >> 16);
    return o.b;
}
__device__ __forceinline__ float bf2f(__bf16 b) {
    union { __bf16 b; unsigned short s; } i; i.b = b;
    union { unsigned u; float f; } o; o.u = ((unsigned)i.s) << 16;
    return o.f;
}

union Frag { v16bf v; v8bf h[2]; };

// ---------------------------------------------------------------------------
// CDNA5 async global->LDS copy (16B per lane) + wait.
// LDS address = low 32 bits of the generic pointer (ISA 10.2: LDS aperture
// addr[31:0] is the wave-relative LDS offset; async op adds LDS_BASE).
// ---------------------------------------------------------------------------
__device__ __forceinline__ void async_copy_b128(void* lds, const void* g) {
    unsigned lo = (unsigned)(uintptr_t)lds;
    asm volatile("global_load_async_to_lds_b128 %0, %1, off"
                 :: "v"(lo), "v"(g) : "memory");
}
__device__ __forceinline__ void wait_async0() {
    asm volatile("s_wait_asynccnt 0x0" ::: "memory");
}

// ---------------------------------------------------------------------------
// Shared epilogue. FLAGS (compile-time): 1=+bias[col], 2=relu,
// 4=clip[1e-6,1], 8=bf16 output. Straight-line code per instantiation.
// ---------------------------------------------------------------------------
template<int FLAGS>
__device__ __forceinline__ void gemm_epilogue(
    const v8f acc[2][2], void* __restrict__ Cout, unsigned ldc,
    const float* __restrict__ bias,
    unsigned mb, unsigned nb, int wr, int wc, int lane)
{
    const unsigned col = lane & 15;
    const unsigned rh  = (lane >> 4) << 3;
    #pragma unroll
    for (int nt = 0; nt < 2; ++nt) {
        const unsigned gc = nb + (unsigned)(wc * 32 + nt * 16) + col;
        float bv = 0.0f;
        if constexpr ((FLAGS & 1) != 0) bv = bias[gc];
        #pragma unroll
        for (int mt = 0; mt < 2; ++mt) {
            const unsigned base = (mb + (unsigned)(wr * 32 + mt * 16) + rh) * ldc + gc;
            #pragma unroll
            for (int j = 0; j < 8; ++j) {
                float vv = acc[mt][nt][j] + bv;
                if constexpr ((FLAGS & 2) != 0) vv = fmaxf(vv, 0.0f);
                if constexpr ((FLAGS & 4) != 0) vv = fminf(fmaxf(vv, 1e-6f), 1.0f);
                if constexpr ((FLAGS & 8) != 0)
                    ((__bf16*)Cout)[base + (unsigned)j * ldc] = f2bf(vv);
                else
                    ((float*)Cout)[base + (unsigned)j * ldc] = vv;
            }
        }
    }
}

// ---------------------------------------------------------------------------
// GEMM (fp32 operands, converted to bf16 while staging). Used once:
// T = A @ features (M=16384, N=256, K=16384) -- bound by the 1 GB A read.
// ---------------------------------------------------------------------------
template<int FLAGS>
__global__ __launch_bounds__(256) void gemm_f32in_wmma(
    const float* __restrict__ A, unsigned lda,
    const float* __restrict__ B, unsigned ldb,
    const float* __restrict__ bias,
    void* __restrict__ Cout, unsigned ldc,
    int K)
{
    __shared__ __bf16 Asl[TBM][LDSA];   // [m][k]
    __shared__ __bf16 Bsl[TBN][LDSB];   // [n][k] (transposed while staging)

    const int t    = threadIdx.x;
    const int lane = t & 31;
    const int wave = t >> 5;
    const int wr   = wave & 3;
    const int wc   = wave >> 2;
    const unsigned mb = blockIdx.y * TBM;
    const unsigned nb = blockIdx.x * TBN;

    v8f acc[2][2];
    #pragma unroll
    for (int i = 0; i < 2; ++i)
        #pragma unroll
        for (int j = 0; j < 2; ++j)
            acc[i][j] = v8f{0.f,0.f,0.f,0.f,0.f,0.f,0.f,0.f};

    const int rsel = lane & 15;
    const int k0   = (lane >> 4) << 3;

    for (int kb = 0; kb < K; kb += TBK) {
        #pragma unroll
        for (int p = 0; p < 4; ++p) {
            const int s  = t + 256 * p;
            const int r  = s >> 3;
            const int k4 = (s & 7) << 2;
            const float* ap = &A[(mb + (unsigned)r) * lda + (unsigned)(kb + k4)];
            if (p == 0) __builtin_prefetch(ap + TBK, 0, 1);  // next k-tile -> global_prefetch_b8
            const float4 fa = *(const float4*)ap;
            v4bf pk;
            pk[0] = f2bf(fa.x); pk[1] = f2bf(fa.y);
            pk[2] = f2bf(fa.z); pk[3] = f2bf(fa.w);
            *(v4bf*)&Asl[r][k4] = pk;
        }
        #pragma unroll
        for (int p = 0; p < 2; ++p) {
            const int s  = t + 256 * p;
            const int k  = s >> 4;
            const int n4 = (s & 15) << 2;
            const float4 fb = *(const float4*)&B[(unsigned)(kb + k) * ldb + nb + (unsigned)n4];
            Bsl[n4 + 0][k] = f2bf(fb.x);
            Bsl[n4 + 1][k] = f2bf(fb.y);
            Bsl[n4 + 2][k] = f2bf(fb.z);
            Bsl[n4 + 3][k] = f2bf(fb.w);
        }
        __syncthreads();

        Frag a[2], b[2];
        #pragma unroll
        for (int mt = 0; mt < 2; ++mt) {
            const int r = wr * 32 + mt * 16 + rsel;
            a[mt].h[0] = *(const v8bf*)&Asl[r][k0];
            a[mt].h[1] = *(const v8bf*)&Asl[r][k0 + 16];
        }
        #pragma unroll
        for (int nt = 0; nt < 2; ++nt) {
            const int n = wc * 32 + nt * 16 + rsel;
            b[nt].h[0] = *(const v8bf*)&Bsl[n][k0];
            b[nt].h[1] = *(const v8bf*)&Bsl[n][k0 + 16];
        }
        #pragma unroll
        for (int mt = 0; mt < 2; ++mt)
            #pragma unroll
            for (int nt = 0; nt < 2; ++nt)
                acc[mt][nt] = __builtin_amdgcn_wmma_f32_16x16x32_bf16(
                    false, a[mt].v, false, b[nt].v,
                    (short)0, acc[mt][nt], false, false);
        __syncthreads();
    }
    gemm_epilogue<FLAGS>(acc, Cout, ldc, bias, mb, nb, wr, wc, lane);
}

// ---------------------------------------------------------------------------
// GEMM (bf16 operands, B pre-transposed: Bt[n][k]).
// Staging = pure 16B global->LDS async copies, double-buffered LDS.
// ---------------------------------------------------------------------------
template<int FLAGS>
__global__ __launch_bounds__(256) void gemm_bf16bt_wmma(
    const __bf16* __restrict__ A, unsigned lda,
    const __bf16* __restrict__ Bt, unsigned ldbt,
    const float* __restrict__ bias,
    void* __restrict__ Cout, unsigned ldc,
    int K)
{
    __shared__ __bf16 Asl[2][TBM][LDSA];
    __shared__ __bf16 Bsl[2][TBN][LDSB];

    const int t    = threadIdx.x;
    const int lane = t & 31;
    const int wave = t >> 5;
    const int wr   = wave & 3;
    const int wc   = wave >> 2;
    const unsigned mb = blockIdx.y * TBM;
    const unsigned nb = blockIdx.x * TBN;

    v8f acc[2][2];
    #pragma unroll
    for (int i = 0; i < 2; ++i)
        #pragma unroll
        for (int j = 0; j < 2; ++j)
            acc[i][j] = v8f{0.f,0.f,0.f,0.f,0.f,0.f,0.f,0.f};

    const int rsel = lane & 15;
    const int k0   = (lane >> 4) << 3;

    auto issue = [&](int buf, int kb) {
        // A tile 128x32 bf16: 512 x 16B chunks, 2 per thread
        #pragma unroll
        for (int p = 0; p < 2; ++p) {
            const int s  = t + 256 * p;
            const int r  = s >> 2;
            const int k8 = (s & 3) << 3;
            async_copy_b128(&Asl[buf][r][k8],
                            &A[(mb + (unsigned)r) * lda + (unsigned)(kb + k8)]);
        }
        // B tile 64x32 bf16: 256 x 16B chunks, 1 per thread
        {
            const int n  = t >> 2;
            const int k8 = (t & 3) << 3;
            async_copy_b128(&Bsl[buf][n][k8],
                            &Bt[(nb + (unsigned)n) * ldbt + (unsigned)(kb + k8)]);
        }
    };

    issue(0, 0);
    const int nk = K / TBK;
    for (int i = 0; i < nk; ++i) {
        const int cur = i & 1;
        wait_async0();
        __syncthreads();
        if (i + 1 < nk) issue(cur ^ 1, (i + 1) * TBK);

        Frag a[2], b[2];
        #pragma unroll
        for (int mt = 0; mt < 2; ++mt) {
            const int r = wr * 32 + mt * 16 + rsel;
            a[mt].h[0] = *(const v8bf*)&Asl[cur][r][k0];
            a[mt].h[1] = *(const v8bf*)&Asl[cur][r][k0 + 16];
        }
        #pragma unroll
        for (int nt = 0; nt < 2; ++nt) {
            const int n = wc * 32 + nt * 16 + rsel;
            b[nt].h[0] = *(const v8bf*)&Bsl[cur][n][k0];
            b[nt].h[1] = *(const v8bf*)&Bsl[cur][n][k0 + 16];
        }
        #pragma unroll
        for (int mt = 0; mt < 2; ++mt)
            #pragma unroll
            for (int nt = 0; nt < 2; ++nt)
                acc[mt][nt] = __builtin_amdgcn_wmma_f32_16x16x32_bf16(
                    false, a[mt].v, false, b[nt].v,
                    (short)0, acc[mt][nt], false, false);
    }
    gemm_epilogue<FLAGS>(acc, Cout, ldc, bias, mb, nb, wr, wc, lane);
}

// ---------------------------------------------------------------------------
// Row L2-normalize (bf16 in/out, rows of 256). One block per row.
// ---------------------------------------------------------------------------
__global__ __launch_bounds__(256) void rownorm_bf16(
    const __bf16* __restrict__ in, __bf16* __restrict__ out)
{
    __shared__ float red[256];
    const unsigned r = blockIdx.x;
    const unsigned t = threadIdx.x;
    const float v = bf2f(in[r * (unsigned)DIMF + t]);
    red[t] = v * v;
    __syncthreads();
    #pragma unroll
    for (int s = 128; s > 0; s >>= 1) {
        if (t < (unsigned)s) red[t] += red[t + s];
        __syncthreads();
    }
    const float inv = 1.0f / fmaxf(sqrtf(red[0]), 1e-8f);
    out[r * (unsigned)DIMF + t] = f2bf(v * inv);
}

// ---------------------------------------------------------------------------
// Weight convert+transpose: W[k][n] fp32 (256x256) -> Wt[n][k] bf16.
// ---------------------------------------------------------------------------
__global__ __launch_bounds__(256) void wt_convert(
    const float* __restrict__ W, __bf16* __restrict__ Wt)
{
    const unsigned k = blockIdx.x;
    const unsigned n = threadIdx.x;
    Wt[n * (unsigned)DIMF + k] = f2bf(W[k * (unsigned)DIMF + n]);
}

// ---------------------------------------------------------------------------
// Host launcher.
// Inputs: A, features, Wg, bg, uW1, ub1, uW2, ub2, iW1, ib1, iW2, ib2
// Workspace (bf16 elems), ~37 MB flat:
//   T   @ 0         (16384x256)     X  @ 4194304  (16384x256)
//   H   @ 8388608   (8192x256)      U  @ 10485760 (8192x256)
//   I   @ 12582912  (8192x256)      Un @ 14680064 (8192x256)
//   In  @ 16777216  (8192x256)      5 x WtT @ 18874368 (256x256 each)
// ---------------------------------------------------------------------------
extern "C" void kernel_launch(void* const* d_in, const int* in_sizes, int n_in,
                              void* d_out, int out_size, void* d_ws, size_t ws_size,
                              hipStream_t stream) {
    (void)in_sizes; (void)n_in; (void)out_size; (void)ws_size;
    const float* A    = (const float*)d_in[0];
    const float* feat = (const float*)d_in[1];
    const float* Wg   = (const float*)d_in[2];
    const float* bg   = (const float*)d_in[3];
    const float* uW1  = (const float*)d_in[4];
    const float* ub1  = (const float*)d_in[5];
    const float* uW2  = (const float*)d_in[6];
    const float* ub2  = (const float*)d_in[7];
    const float* iW1  = (const float*)d_in[8];
    const float* ib1  = (const float*)d_in[9];
    const float* iW2  = (const float*)d_in[10];
    const float* ib2  = (const float*)d_in[11];
    float*  out = (float*)d_out;
    __bf16* wb  = (__bf16*)d_ws;

    __bf16* T    = wb;
    __bf16* X    = wb + (size_t)4194304;
    __bf16* H    = wb + (size_t)8388608;
    __bf16* U    = wb + (size_t)10485760;
    __bf16* I    = wb + (size_t)12582912;
    __bf16* Un   = wb + (size_t)14680064;
    __bf16* In   = wb + (size_t)16777216;
    __bf16* WgT  = wb + (size_t)18874368;
    __bf16* uW1T = WgT  + 65536;
    __bf16* uW2T = uW1T + 65536;
    __bf16* iW1T = uW2T + 65536;
    __bf16* iW2T = iW1T + 65536;

    const dim3 blk(256, 1, 1);

    // 0) convert + transpose the 5 weight matrices to bf16 [n][k]
    wt_convert<<<dim3(DIMF), blk, 0, stream>>>(Wg,  WgT);
    wt_convert<<<dim3(DIMF), blk, 0, stream>>>(uW1, uW1T);
    wt_convert<<<dim3(DIMF), blk, 0, stream>>>(uW2, uW2T);
    wt_convert<<<dim3(DIMF), blk, 0, stream>>>(iW1, iW1T);
    wt_convert<<<dim3(DIMF), blk, 0, stream>>>(iW2, iW2T);

    // 1) T = A @ features (fp32-convert path), bf16 out
    gemm_f32in_wmma<8><<<dim3(DIMF / TBN, NTOT / TBM), blk, 0, stream>>>(
        A, NTOT, feat, DIMF, nullptr, T, DIMF, NTOT);
    // 2) X = relu(T @ Wg + bg), bf16 out
    gemm_bf16bt_wmma<1 | 2 | 8><<<dim3(DIMF / TBN, NTOT / TBM), blk, 0, stream>>>(
        T, DIMF, WgT, DIMF, bg, X, DIMF, DIMF);
    // 3) Hu = relu(X[:8192] @ uW1 + ub1)
    gemm_bf16bt_wmma<1 | 2 | 8><<<dim3(DIMF / TBN, HALF / TBM), blk, 0, stream>>>(
        X, DIMF, uW1T, DIMF, ub1, H, DIMF, DIMF);
    // 4) U = Hu @ uW2 + ub2
    gemm_bf16bt_wmma<1 | 8><<<dim3(DIMF / TBN, HALF / TBM), blk, 0, stream>>>(
        H, DIMF, uW2T, DIMF, ub2, U, DIMF, DIMF);
    // 5) Hi = relu(X[8192:] @ iW1 + ib1)
    gemm_bf16bt_wmma<1 | 2 | 8><<<dim3(DIMF / TBN, HALF / TBM), blk, 0, stream>>>(
        X + (size_t)HALF * DIMF, DIMF, iW1T, DIMF, ib1, H, DIMF, DIMF);
    // 6) I = Hi @ iW2 + ib2
    gemm_bf16bt_wmma<1 | 8><<<dim3(DIMF / TBN, HALF / TBM), blk, 0, stream>>>(
        H, DIMF, iW2T, DIMF, ib2, I, DIMF, DIMF);
    // 7) normalize rows (both stay row-major: In IS the Bt operand)
    rownorm_bf16<<<dim3(HALF), blk, 0, stream>>>(U, Un);
    rownorm_bf16<<<dim3(HALF), blk, 0, stream>>>(I, In);
    // 8) sim = clip(Un @ In^T, 1e-6, 1.0) -> fp32 d_out
    gemm_bf16bt_wmma<4><<<dim3(HALF / TBN, HALF / TBM), blk, 0, stream>>>(
        Un, DIMF, In, DIMF, nullptr, out, HALF, DIMF);
}
